// SparseMoeBlock_58480274702424
// MI455X (gfx1250) — matrix operand order repaired and checked by reading, not compiled, not run on hardware
//
#include <hip/hip_runtime.h>
#include <hip/hip_bf16.h>
#include <stdint.h>

#define B_ 16
#define S_ 1024
#define D_ 1024
#define E_ 8
#define IM_ 2048
#define KCAP_ 256
#define UNCOND_LABEL_ 1000

typedef __attribute__((ext_vector_type(16))) __bf16 v16bf;
typedef __attribute__((ext_vector_type(8)))  float  v8f;
typedef unsigned short u16;
typedef unsigned int   u32;

// ---------------- helpers ----------------

static __device__ __forceinline__ u16 f2bf(float f) {
  union { float f; u32 u; } x; x.f = f;
  u32 u = x.u;
  u32 r = (u + 0x7fffu + ((u >> 16) & 1u)) >> 16;  // round-to-nearest-even
  return (u16)r;
}

union FragBF { uint4 q[2]; v16bf v; };

static __device__ __forceinline__ v8f wmma_bf16(v16bf a, v16bf b, v8f c) {
  return __builtin_amdgcn_wmma_f32_16x16x32_bf16(false, a, false, b, (short)0, c, false, false);
}

// B fragment from per-lane base (row n, K elems/row); j selects 16-row sub-tile.
template <int K>
static __device__ __forceinline__ void loadB(FragBF& f, const u16* base, int j) {
  f.q[0] = *(const uint4*)(base + j * 16 * K);
  f.q[1] = *(const uint4*)(base + j * 16 * K + 8);
}

static __device__ __forceinline__ float silu(float x) {
  return x / (1.0f + __expf(-x));
}

static __device__ __forceinline__ float block_reduce_sum(float v, float* sm, int nthreads) {
  for (int o = 16; o > 0; o >>= 1) v += __shfl_down(v, o, 32);
  int lane = threadIdx.x & 31, w = threadIdx.x >> 5;
  __syncthreads();
  if (lane == 0) sm[w] = v;
  __syncthreads();
  float r = 0.f;
  int nw = nthreads >> 5;
  if ((int)threadIdx.x < nw) r = sm[threadIdx.x];
  if (w == 0) {
    for (int o = 4; o > 0; o >>= 1) r += __shfl_down(r, o, 32);
    if (lane == 0) sm[0] = r;
  }
  __syncthreads();
  return sm[0];
}

static __device__ __forceinline__ float block_reduce_max(float v, float* sm, int nthreads) {
  for (int o = 16; o > 0; o >>= 1) v = fmaxf(v, __shfl_down(v, o, 32));
  int lane = threadIdx.x & 31, w = threadIdx.x >> 5;
  __syncthreads();
  if (lane == 0) sm[w] = v;
  __syncthreads();
  float r = -3.0e38f;
  int nw = nthreads >> 5;
  if ((int)threadIdx.x < nw) r = sm[threadIdx.x];
  if (w == 0) {
    for (int o = 4; o > 0; o >>= 1) r = fmaxf(r, __shfl_down(r, o, 32));
    if (lane == 0) sm[0] = r;
  }
  __syncthreads();
  return sm[0];
}

// ---------------- router ----------------

__global__ void normc_kernel(const float* __restrict__ cc, float* __restrict__ cn) {
  __shared__ float sm[8];
  int e = blockIdx.x, tid = threadIdx.x;  // 256 threads
  float vals[4]; float s2 = 0.f;
  for (int i = 0; i < 4; i++) { vals[i] = cc[(size_t)e * D_ + tid + i * 256]; s2 += vals[i] * vals[i]; }
  s2 = block_reduce_sum(s2, sm, 256);
  float rinv = 1.0f / fmaxf(sqrtf(s2), 1e-12f);
  for (int i = 0; i < 4; i++) cn[(size_t)e * D_ + tid + i * 256] = vals[i] * rinv;
}

__global__ void cos_kernel(const float* __restrict__ hs, const float* __restrict__ cn,
                           float* __restrict__ probs) {
  __shared__ float sm[8];
  int token = blockIdx.x;                // b*S + s
  int b = token >> 10, s = token & 1023;
  const float* x = hs + (size_t)token * D_;
  int tid = threadIdx.x;                 // 128 threads
  float x2 = 0.f, dot[E_];
  for (int e = 0; e < E_; e++) dot[e] = 0.f;
  for (int i = 0; i < 8; i++) {
    int d = tid + i * 128;
    float xv = x[d];
    x2 += xv * xv;
#pragma unroll
    for (int e = 0; e < E_; e++) dot[e] += xv * cn[(size_t)e * D_ + d];
  }
  x2 = block_reduce_sum(x2, sm, 128);
  float rinv = 1.0f / fmaxf(sqrtf(x2), 1e-12f);
  for (int e = 0; e < E_; e++) {
    float de = block_reduce_sum(dot[e], sm, 128);
    if (tid == 0) probs[((size_t)b * E_ + e) * S_ + s] = de * rinv;
  }
}

__global__ void softmax_kernel(float* __restrict__ probs) {
  __shared__ float sm[9];
  int g = blockIdx.x, tid = threadIdx.x;  // 256 threads, S_=1024
  float* p = probs + (size_t)g * S_;
  float vals[4]; float mx = -3.0e38f;
  for (int i = 0; i < 4; i++) { vals[i] = p[tid + i * 256]; mx = fmaxf(mx, vals[i]); }
  mx = block_reduce_max(mx, sm, 256);
  float sum = 0.f;
  for (int i = 0; i < 4; i++) { vals[i] = __expf(vals[i] - mx); sum += vals[i]; }
  sum = block_reduce_sum(sum, sm, 256);
  float inv = 1.0f / sum;
  for (int i = 0; i < 4; i++) p[tid + i * 256] = vals[i] * inv;
}

// radix-select top-KCAP per (b,e); order irrelevant (only scatter-added sums matter)
__global__ void topk_kernel(const float* __restrict__ probs, int* __restrict__ idxb,
                            float* __restrict__ gateb) {
  __shared__ int cnt, cg, ce;
  int g = blockIdx.x, tid = threadIdx.x;  // 256 threads
  const float* p = probs + (size_t)g * S_;
  u32 v[4];
  for (int i = 0; i < 4; i++) { union { float f; u32 u; } x; x.f = p[tid + i * 256]; v[i] = x.u; }
  // positive floats: uint order == float order
  u32 prefix = 0;
  for (int bit = 31; bit >= 0; --bit) {
    u32 cand = prefix | (1u << bit);
    if (tid == 0) cnt = 0;
    __syncthreads();
    int c = 0;
    for (int i = 0; i < 4; i++) c += (v[i] >= cand);
    if (c) atomicAdd(&cnt, c);
    __syncthreads();
    int total = cnt;
    __syncthreads();
    if (total >= KCAP_) prefix = cand;
  }
  u32 thr = prefix;  // count(v>=thr) >= K, count(v>thr) < K
  if (tid == 0) cnt = 0;
  __syncthreads();
  { int c = 0; for (int i = 0; i < 4; i++) c += (v[i] > thr); if (c) atomicAdd(&cnt, c); }
  __syncthreads();
  int ngt = cnt;
  if (tid == 0) { cg = 0; ce = 0; }
  __syncthreads();
  for (int i = 0; i < 4; i++) {
    int s = tid + i * 256;
    if (v[i] > thr) {
      int pos = atomicAdd(&cg, 1);
      idxb[(size_t)g * KCAP_ + pos] = s;
      union { u32 u; float f; } x; x.u = v[i];
      gateb[(size_t)g * KCAP_ + pos] = x.f;
    }
  }
  for (int i = 0; i < 4; i++) {
    int s = tid + i * 256;
    if (v[i] == thr) {
      int pos = ngt + atomicAdd(&ce, 1);
      if (pos < KCAP_) {
        idxb[(size_t)g * KCAP_ + pos] = s;
        union { u32 u; float f; } x; x.u = thr;
        gateb[(size_t)g * KCAP_ + pos] = x.f;
      }
    }
  }
}

// ---------------- data movement / conversion ----------------

__global__ void zero_kernel(float4* __restrict__ p) {
  size_t t = (size_t)blockIdx.x * blockDim.x + threadIdx.x;
  float4 z; z.x = z.y = z.z = z.w = 0.f;
  p[t] = z;
}

__global__ void cvt_bf16_kernel(const float* __restrict__ in, u16* __restrict__ out) {
  size_t t = (size_t)blockIdx.x * blockDim.x + threadIdx.x;
  float4 f = *(const float4*)(in + t * 4);
  ushort4 o; o.x = f2bf(f.x); o.y = f2bf(f.y); o.z = f2bf(f.z); o.w = f2bf(f.w);
  *(ushort4*)(out + t * 4) = o;
}

__global__ void gather_kernel(const float* __restrict__ hs, const int* __restrict__ idxb,
                              u16* __restrict__ Xr) {
  size_t t = (size_t)blockIdx.x * blockDim.x + threadIdx.x;  // (g,k,d4)
  int d4 = (int)(t & 255);          // D/4 = 256
  size_t r = t >> 8;
  int k = (int)(r & 255);           // KCAP = 256
  int g = (int)(r >> 8);            // B*E = 128
  int b = g >> 3;
  int s = idxb[(size_t)g * KCAP_ + k];
  float4 f = *(const float4*)(hs + ((size_t)(b * S_ + s)) * D_ + d4 * 4);
  ushort4 o; o.x = f2bf(f.x); o.y = f2bf(f.y); o.z = f2bf(f.z); o.w = f2bf(f.w);
  *(ushort4*)(Xr + ((size_t)g * KCAP_ + k) * D_ + d4 * 4) = o;
}

// fp32 (R x C) -> bf16 (C x R) transpose-convert, batched by blockIdx.z
__global__ void transpose_bf16(const float* __restrict__ in, u16* __restrict__ out,
                               int R, int C, size_t in_bs, size_t out_bs) {
  __shared__ float tile[32][33];
  const float* inb = in + (size_t)blockIdx.z * in_bs;
  u16* outb = out + (size_t)blockIdx.z * out_bs;
  int c0 = blockIdx.x * 32, r0 = blockIdx.y * 32;
  int tx = threadIdx.x, ty = threadIdx.y;  // 32 x 8
  for (int i = 0; i < 4; i++) {
    int r = r0 + ty + i * 8;
    tile[ty + i * 8][tx] = inb[(size_t)r * C + c0 + tx];
  }
  __syncthreads();
  for (int i = 0; i < 4; i++) {
    int c = c0 + ty + i * 8;
    outb[(size_t)c * R + r0 + tx] = f2bf(tile[tx][ty + i * 8]);
  }
}

// ---------------- WMMA GEMM kernels ----------------
// Block = 256 threads (8 waves). K is a compile-time leading dimension so all
// B sub-tile addresses fold into instruction offsets from one base pointer.
// All B fragments of a k-step are loaded into DISTINCT registers before any
// WMMA issues. __launch_bounds__(256, 1) relaxes the backend's waves-per-EU
// register cap so the scheduler can keep the fragments live (partial
// s_wait_loadcnt instead of full drains before every WMMA).

// H = bf16( silu(A @ Wg) * (A @ Wu) ); wave tile 16M x 64N (A reused 8x)
template <int K>
__global__ __launch_bounds__(256, 1) void gateup_wmma(
    const u16* __restrict__ A, const u16* __restrict__ WgT,
    const u16* __restrict__ WuT, u16* __restrict__ H,
    int I, long a_stride, long w_stride, long h_stride) {
  int lane = threadIdx.x & 31;
  int wv = threadIdx.x >> 5;
  int g = blockIdx.z;
  int half = lane >> 4, nc = lane & 15;
  int m0 = (blockIdx.x * 8 + wv) * 16;
  int n0 = blockIdx.y * 64;

  const u16* aP  = A   + (size_t)g * a_stride + (size_t)(m0 + nc) * K + half * 16;
  const u16* bgP = WgT + (size_t)(g % E_) * w_stride + (size_t)(n0 + nc) * K + half * 16;
  const u16* buP = WuT + (size_t)(g % E_) * w_stride + (size_t)(n0 + nc) * K + half * 16;
  u16* Hg = H + (size_t)g * h_stride;

  v8f accg[4], accu[4];
#pragma unroll
  for (int j = 0; j < 4; j++) { accg[j] = {}; accu[j] = {}; }

  for (int k0 = 0; k0 < K; k0 += 32) {
    FragBF a;
    a.q[0] = *(const uint4*)(aP);
    a.q[1] = *(const uint4*)(aP + 8);
    __builtin_prefetch((const void*)(aP + 128), 0, 1);
    FragBF bg[4], bu[4];
#pragma unroll
    for (int j = 0; j < 4; j++) { loadB<K>(bg[j], bgP, j); loadB<K>(bu[j], buP, j); }
#pragma unroll
    for (int j = 0; j < 4; j++) {
      accg[j] = wmma_bf16(a.v, bg[j].v, accg[j]);
      accu[j] = wmma_bf16(a.v, bu[j].v, accu[j]);
    }
    aP += 32; bgP += 32; buP += 32;
  }

#pragma unroll
  for (int j = 0; j < 4; j++) {
#pragma unroll
    for (int r = 0; r < 8; r++) {
      float gt = accg[j][r], up = accu[j][r];
      int mm = m0 + r + 8 * half;
      Hg[(size_t)mm * I + n0 + j * 16 + nc] = f2bf(silu(gt) * up);
    }
  }
}

// O = H @ Wd; wave tile 16M x 128N (A reused 8x). Epilogue:
// MODE 0: routed  -> atomicAdd gating[m]*val into cond[b, idx[m], col]
// MODE 1: shared  -> store val to out[token, col]
// MODE 2: uncond  -> if labels[b]==1000: out[token, col] += val
template <int K, int MODE>
__global__ __launch_bounds__(256, 1) void down_wmma(
    const u16* __restrict__ H, const u16* __restrict__ WdT,
    long h_stride, long w_stride,
    const float* __restrict__ gating, const int* __restrict__ idxb,
    const long long* __restrict__ labels,
    float* __restrict__ cond, float* __restrict__ outp) {
  int lane = threadIdx.x & 31;
  int wv = threadIdx.x >> 5;
  int g = blockIdx.z;
  int half = lane >> 4, nc = lane & 15;
  int m0 = (blockIdx.x * 8 + wv) * 16;
  int n0 = blockIdx.y * 128;

  const u16* aP = H   + (size_t)g * h_stride + (size_t)(m0 + nc) * K + half * 16;
  const u16* bP = WdT + (size_t)(g % E_) * w_stride + (size_t)(n0 + nc) * K + half * 16;

  v8f acc[8];
#pragma unroll
  for (int j = 0; j < 8; j++) acc[j] = {};

  for (int k0 = 0; k0 < K; k0 += 32) {
    FragBF a;
    a.q[0] = *(const uint4*)(aP);
    a.q[1] = *(const uint4*)(aP + 8);
    __builtin_prefetch((const void*)(aP + 128), 0, 1);
    FragBF b[8];
#pragma unroll
    for (int j = 0; j < 8; j++) loadB<K>(b[j], bP, j);
#pragma unroll
    for (int j = 0; j < 8; j++) acc[j] = wmma_bf16(a.v, b[j].v, acc[j]);
    aP += 32; bP += 32;
  }

#pragma unroll
  for (int j = 0; j < 8; j++) {
    int col = n0 + j * 16 + nc;
#pragma unroll
    for (int r = 0; r < 8; r++) {
      float val = acc[j][r];
      int mm = m0 + r + 8 * half;
      if (MODE == 0) {
        int b = g >> 3;
        float w = gating[(size_t)g * KCAP_ + mm];
        int tok = idxb[(size_t)g * KCAP_ + mm];
        atomicAdd(&cond[((size_t)b * S_ + tok) * D_ + col], val * w);
      } else if (MODE == 1) {
        outp[(size_t)mm * D_ + col] = val;
      } else {
        int b = mm / S_;
        if (labels[b] == UNCOND_LABEL_) outp[(size_t)mm * D_ + col] += val;
      }
    }
  }
}

__global__ void combine_kernel(const float4* __restrict__ cond, float4* __restrict__ outp,
                               const long long* __restrict__ labels) {
  size_t t = (size_t)blockIdx.x * blockDim.x + threadIdx.x;  // B*S*D/4
  int b = (int)(t / (size_t)(S_ * D_ / 4));
  if (labels[b] != UNCOND_LABEL_) {
    float4 o = outp[t]; float4 c = cond[t];
    o.x += c.x; o.y += c.y; o.z += c.z; o.w += c.w;
    outp[t] = o;
  }
}

// ---------------- launcher ----------------

extern "C" void kernel_launch(void* const* d_in, const int* in_sizes, int n_in,
                              void* d_out, int out_size, void* d_ws, size_t ws_size,
                              hipStream_t stream) {
  const float* hs = (const float*)d_in[0];
  const long long* labels = (const long long*)d_in[1];
  const float* cc = (const float*)d_in[2];
  const float* Wg = (const float*)d_in[3];
  const float* Wu = (const float*)d_in[4];
  const float* Wd = (const float*)d_in[5];
  const float* uWg = (const float*)d_in[6];
  const float* uWu = (const float*)d_in[7];
  const float* uWd = (const float*)d_in[8];
  const float* sWg = (const float*)d_in[9];
  const float* sWu = (const float*)d_in[10];
  const float* sWd = (const float*)d_in[11];
  float* out = (float*)d_out;

  char* p = (char*)d_ws;
  auto alloc = [&](size_t bytes) -> void* {
    void* r = (void*)p;
    p += (bytes + 255) & ~(size_t)255;
    return r;
  };
  float* cn    = (float*)alloc((size_t)E_ * D_ * 4);
  float* probs = (float*)alloc((size_t)B_ * E_ * S_ * 4);
  int*   idxb  = (int*)alloc((size_t)B_ * E_ * KCAP_ * 4);
  float* gateb = (float*)alloc((size_t)B_ * E_ * KCAP_ * 4);
  u16* xb   = (u16*)alloc((size_t)B_ * S_ * D_ * 2);
  u16* Xr   = (u16*)alloc((size_t)B_ * E_ * KCAP_ * D_ * 2);
  u16* WgT  = (u16*)alloc((size_t)E_ * IM_ * D_ * 2);
  u16* WuT  = (u16*)alloc((size_t)E_ * IM_ * D_ * 2);
  u16* WdT  = (u16*)alloc((size_t)E_ * D_ * IM_ * 2);
  u16* sWgT = (u16*)alloc((size_t)IM_ * D_ * 2);
  u16* sWuT = (u16*)alloc((size_t)IM_ * D_ * 2);
  u16* sWdT = (u16*)alloc((size_t)D_ * IM_ * 2);
  u16* uWgT = (u16*)alloc((size_t)IM_ * D_ * 2);
  u16* uWuT = (u16*)alloc((size_t)IM_ * D_ * 2);
  u16* uWdT = (u16*)alloc((size_t)D_ * IM_ * 2);
  u16* Hr   = (u16*)alloc((size_t)B_ * E_ * KCAP_ * IM_ * 2);
  u16* Hs   = (u16*)alloc((size_t)B_ * S_ * IM_ * 2);
  float* cond = (float*)alloc((size_t)B_ * S_ * D_ * 4);

  dim3 tblk(32, 8);

  // zero cond scatter buffer
  zero_kernel<<<16384, 256, 0, stream>>>((float4*)cond);
  // router
  normc_kernel<<<E_, 256, 0, stream>>>(cc, cn);
  cos_kernel<<<B_ * S_, 128, 0, stream>>>(hs, cn, probs);
  softmax_kernel<<<B_ * E_, 256, 0, stream>>>(probs);
  topk_kernel<<<B_ * E_, 256, 0, stream>>>(probs, idxb, gateb);
  // conversions
  cvt_bf16_kernel<<<16384, 256, 0, stream>>>(hs, xb);
  gather_kernel<<<32768, 256, 0, stream>>>(hs, idxb, Xr);
  // weight transpose-converts (B^T layout => contiguous K per output column)
  transpose_bf16<<<dim3(IM_ / 32, D_ / 32, E_), tblk, 0, stream>>>(Wg, WgT, D_, IM_, (size_t)D_ * IM_, (size_t)IM_ * D_);
  transpose_bf16<<<dim3(IM_ / 32, D_ / 32, E_), tblk, 0, stream>>>(Wu, WuT, D_, IM_, (size_t)D_ * IM_, (size_t)IM_ * D_);
  transpose_bf16<<<dim3(D_ / 32, IM_ / 32, E_), tblk, 0, stream>>>(Wd, WdT, IM_, D_, (size_t)IM_ * D_, (size_t)D_ * IM_);
  transpose_bf16<<<dim3(IM_ / 32, D_ / 32, 1), tblk, 0, stream>>>(sWg, sWgT, D_, IM_, 0, 0);
  transpose_bf16<<<dim3(IM_ / 32, D_ / 32, 1), tblk, 0, stream>>>(sWu, sWuT, D_, IM_, 0, 0);
  transpose_bf16<<<dim3(D_ / 32, IM_ / 32, 1), tblk, 0, stream>>>(sWd, sWdT, IM_, D_, 0, 0);
  transpose_bf16<<<dim3(IM_ / 32, D_ / 32, 1), tblk, 0, stream>>>(uWg, uWgT, D_, IM_, 0, 0);
  transpose_bf16<<<dim3(IM_ / 32, D_ / 32, 1), tblk, 0, stream>>>(uWu, uWuT, D_, IM_, 0, 0);
  transpose_bf16<<<dim3(D_ / 32, IM_ / 32, 1), tblk, 0, stream>>>(uWd, uWdT, IM_, D_, 0, 0);

  // routed experts: M=256 per (b,e) group, K=D, I=IM
  gateup_wmma<D_><<<dim3(2, IM_ / 64, B_ * E_), 256, 0, stream>>>(
      Xr, WgT, WuT, Hr, IM_,
      (long)KCAP_ * D_, (long)IM_ * D_, (long)KCAP_ * IM_);
  down_wmma<IM_, 0><<<dim3(2, D_ / 128, B_ * E_), 256, 0, stream>>>(
      Hr, WdT, (long)KCAP_ * IM_, (long)D_ * IM_,
      gateb, idxb, labels, cond, nullptr);

  // shared expert over all tokens: M = B*S = 16384
  gateup_wmma<D_><<<dim3(128, IM_ / 64, 1), 256, 0, stream>>>(
      xb, sWgT, sWuT, Hs, IM_, 0, 0, 0);
  down_wmma<IM_, 1><<<dim3(128, D_ / 128, 1), 256, 0, stream>>>(
      Hs, sWdT, 0, 0, nullptr, nullptr, labels, nullptr, out);

  // uncond expert, added only where labels == 1000
  gateup_wmma<D_><<<dim3(128, IM_ / 64, 1), 256, 0, stream>>>(
      xb, uWgT, uWuT, Hs, IM_, 0, 0, 0);
  down_wmma<IM_, 2><<<dim3(128, D_ / 128, 1), 256, 0, stream>>>(
      Hs, uWdT, 0, 0, nullptr, nullptr, labels, nullptr, out);

  // add routed cond output where labels != 1000
  combine_kernel<<<16384, 256, 0, stream>>>((const float4*)cond, (float4*)out, labels);
}
